// CustomMultiheadAttention_21157008900430
// MI455X (gfx1250) — matrix operand order repaired and checked by reading, not compiled, not run on hardware
//
#include <hip/hip_runtime.h>
#include <hip/hip_bf16.h>

typedef __attribute__((ext_vector_type(16))) _Float16 v16h;
typedef __attribute__((ext_vector_type(8)))  float    v8f;
typedef __attribute__((ext_vector_type(8)))  unsigned int v8u;
typedef __attribute__((ext_vector_type(4)))  unsigned int v4u;
typedef __attribute__((ext_vector_type(8)))  int      v8i;
typedef __attribute__((ext_vector_type(4)))  int      v4i;

#define E_DIM 1024
#define S_LEN 2048
#define BATCH 2
#define HEADS 16
#define HDIM  64
#define MTOT  (BATCH * S_LEN)   // 4096

// LDS row strides (16B-aligned rows so TDM row padding lands exactly)
#define LDK 40   // 64x32 f16 tiles: 64B row + 16B pad
#define LDA 72   // 64x64 f16 tiles: 128B row + 16B pad

// ---------- helpers ----------
__device__ __forceinline__ unsigned short f2h(float f) {
    _Float16 h = (_Float16)f;
    return __builtin_bit_cast(unsigned short, h);
}

__device__ __forceinline__ unsigned lds_off(const void* p) {
    return (unsigned)(unsigned long long)p;   // low 32 bits = LDS byte offset
}

// Tensor Data Mover: 2D f16 tile load Global -> LDS with per-row padding.
// D# group0/group1 packed per CDNA5 ISA 8.3/8.4. count=1, type=2 (image),
// data_size=2B, pad_enable, tight tensor dims (tiles never go OOB here).
// This toolchain exposes the 6-arg builtin: (v4u, v8i, v4i, v4i, v8i, i32).
__device__ __forceinline__ void tdm_load_2d_f16(
    unsigned lds_byte_off, const unsigned short* gptr,
    unsigned tile_d0, unsigned tile_d1,
    unsigned stride0_elems,
    unsigned pad_interval_code,   // (bytesStored>>3) >= 2^code -> pad
    unsigned pad_amount_code) {   // pad DWORDs = code+1
    unsigned long long ga = (unsigned long long)gptr;
    v4u g0;
    g0[0] = 1u;                                   // count=1 (valid descriptor)
    g0[1] = lds_byte_off;                         // lds_addr
    g0[2] = (unsigned)ga;                         // global_addr[31:0]
    g0[3] = ((unsigned)(ga >> 32) & 0x1FFFFFFu)   // global_addr[56:32]
            | (2u << 30);                         // type=2
    const unsigned w0 = (1u << 16)                // data_size = 2 bytes
                      | (1u << 20)                // pad_enable
                      | (pad_interval_code << 22)
                      | (pad_amount_code << 25);
    v8i g1;
    g1[0] = (int)w0;                              // mask=0 | size | pad cfg
    g1[1] = (int)((tile_d0 & 0xFFFFu) << 16);     // tensor_dim0[15:0] @63:48
    g1[2] = (int)((tile_d1 & 0xFFFFu) << 16);     // dim0 hi=0 | tensor_dim1 lo
    g1[3] = (int)((tile_d0 & 0xFFFFu) << 16);     // dim1 hi=0 | tile_dim0
    g1[4] = (int)(tile_d1 & 0xFFFFu);             // tile_dim1 | tile_dim2=0
    g1[5] = (int)stride0_elems;                   // tensor_dim0_stride lo
    g1[6] = 0;
    g1[7] = 0;
    v4i z4 = {0, 0, 0, 0};
    v8i z8 = {0, 0, 0, 0, 0, 0, 0, 0};
    __builtin_amdgcn_tensor_load_to_lds(g0, g1, z4, z4, z8, 0);
}

// A-fragment (16x32 f16): lane = M; VGPR v holds K pair
// k = (lane>>4)*16 + v*2
__device__ __forceinline__ v16h load_a_frag(const unsigned short* lds, int ld, int lane) {
    int m = lane & 15;
    int half = lane >> 4;
    v8u r;
#pragma unroll
    for (int v = 0; v < 8; ++v) {
        int k = (half << 4) + (v << 1);
        const unsigned short* p = lds + m * ld + k;
        r[v] = (unsigned)p[0] | ((unsigned)p[1] << 16);
    }
    return __builtin_bit_cast(v16h, r);
}

// B-fragment (32x16 f16), logical B[k][n] = lds[n*ld + k] (transposed read)
__device__ __forceinline__ v16h load_b_fragT(const unsigned short* lds, int ld, int lane) {
    int n = lane & 15;
    int kb = (lane >> 4) << 4;
    v8u r;
#pragma unroll
    for (int v = 0; v < 8; ++v) {
        const unsigned short* p = lds + n * ld + kb + 2 * v;
        r[v] = (unsigned)p[0] | ((unsigned)p[1] << 16);
    }
    return __builtin_bit_cast(v16h, r);
}

// B-fragment (32x16 f16), logical B[k][n] = lds[k*ld + n]
__device__ __forceinline__ v16h load_b_fragN(const unsigned short* lds, int ld, int lane) {
    int n = lane & 15;
    int kb = (lane >> 4) << 4;
    v8u r;
#pragma unroll
    for (int v = 0; v < 8; ++v) {
        int k = kb + 2 * v;
        r[v] = (unsigned)lds[k * ld + n] | ((unsigned)lds[(k + 1) * ld + n] << 16);
    }
    return __builtin_bit_cast(v16h, r);
}

__device__ __forceinline__ v8f wmma16(v16h a, v16h b, v8f c) {
    return __builtin_amdgcn_wmma_f32_16x16x32_f16(false, a, false, b, (short)0, c, false, false);
}

// ================= Kernel 0: fp32 -> f16 convert (weights) =================
__global__ __launch_bounds__(256) void cvt_f16_kernel(
    const float* __restrict__ in, unsigned short* __restrict__ out, int n4) {
    const int i = blockIdx.x * blockDim.x + threadIdx.x;
    if (i < n4) {
        float4 f = *(const float4*)(in + (size_t)i * 4);
        unsigned short* p = out + (size_t)i * 4;
        p[0] = f2h(f.x); p[1] = f2h(f.y); p[2] = f2h(f.z); p[3] = f2h(f.w);
    }
}

// ================= Kernel 1: QKV projection =================
// Out[b,h,s,d] (f16) = sum_k X[b,s,k] * W[h*64+d, k] + bias[h*64+d]
// BM=64, BN=64, BK=32; 256 threads = 8 waves (4 x 2). B tile via TDM.
__global__ __launch_bounds__(256) void proj_qkv_kernel(
    const float* __restrict__ X, const unsigned short* __restrict__ W16,
    const float* __restrict__ bias, unsigned short* __restrict__ Out) {
    __shared__ __align__(16) unsigned short As[64 * LDK];
    __shared__ __align__(16) unsigned short Bs[64 * LDK];
    const int m0 = blockIdx.x * 64;
    const int n0 = blockIdx.y * 64;
    const int t = threadIdx.x;
    const int lane = t & 31;
    const int wave = t >> 5;
    const int wm = wave >> 1;      // 0..3
    const int wn = wave & 1;       // 0..1

    v8f acc0 = {}, acc1 = {};
    for (int k0 = 0; k0 < E_DIM; k0 += 32) {
        __syncthreads();
        if (wave == 0) {   // B tile: 64 rows x 32 cols f16 (TDM, padded rows)
            tdm_load_2d_f16(lds_off(Bs), W16 + (size_t)n0 * E_DIM + k0,
                            32, 64, E_DIM, 3, 3);
        }
        {   // A tile: fp32 -> f16 convert stage
            const int r  = t >> 3;          // 0..31
            const int c4 = (t & 7) * 4;     // 0..28
#pragma unroll
            for (int i = 0; i < 2; ++i) {
                const int row = r + i * 32;
                float4 fa = *(const float4*)(X + (size_t)(m0 + row) * E_DIM + k0 + c4);
                unsigned short* pa = As + row * LDK + c4;
                pa[0] = f2h(fa.x); pa[1] = f2h(fa.y); pa[2] = f2h(fa.z); pa[3] = f2h(fa.w);
            }
        }
        if (wave == 0) __builtin_amdgcn_s_wait_tensorcnt(0);
        __syncthreads();
        v16h a  = load_a_frag(As + (wm * 16) * LDK, LDK, lane);
        v16h b0 = load_b_fragT(Bs + (wn * 32) * LDK, LDK, lane);
        v16h b1 = load_b_fragT(Bs + (wn * 32 + 16) * LDK, LDK, lane);
        acc0 = wmma16(a, b0, acc0);
        acc1 = wmma16(a, b1, acc1);
    }
    const int half = lane >> 4;
    const int nl = lane & 15;
#pragma unroll
    for (int v = 0; v < 8; ++v) {
        const int m = m0 + wm * 16 + v + 8 * half;
        const int b = m >> 11;            // /S_LEN
        const int s = m & (S_LEN - 1);
        {
            const int n = n0 + wn * 32 + nl;
            const float val = acc0[v] + bias[n];
            const int h = n >> 6, d = n & 63;
            Out[(((size_t)(b * HEADS + h) * S_LEN + s) * HDIM) + d] = f2h(val);
        }
        {
            const int n = n0 + wn * 32 + 16 + nl;
            const float val = acc1[v] + bias[n];
            const int h = n >> 6, d = n & 63;
            Out[(((size_t)(b * HEADS + h) * S_LEN + s) * HDIM) + d] = f2h(val);
        }
    }
}

// ================= Kernel 2: flash attention (causal) =================
// 128 threads = 4 waves; 64 queries (16/wave) x 64-key tiles. Q/K/V via TDM.
__global__ __launch_bounds__(128) void flash_attn_kernel(
    const unsigned short* __restrict__ Qh, const unsigned short* __restrict__ Kh,
    const unsigned short* __restrict__ Vh, unsigned short* __restrict__ CTX) {
    __shared__ __align__(16) unsigned short Qs[64 * LDA];
    __shared__ __align__(16) unsigned short Ks[64 * LDA];
    __shared__ __align__(16) unsigned short Vs[64 * LDA];
    __shared__ __align__(16) unsigned short Ps[4 * 16 * LDA];

    const int t = threadIdx.x;
    const int lane = t & 31;
    const int wave = t >> 5;
    const int q0 = blockIdx.x * 64;
    const int bh = blockIdx.y;                 // b*HEADS + h
    const size_t base = (size_t)bh * S_LEN * HDIM;
    const unsigned short* Qb = Qh + base;
    const unsigned short* Kb = Kh + base;
    const unsigned short* Vb = Vh + base;

    if (wave == 0) {   // Q tile 64x64 f16 via TDM
        tdm_load_2d_f16(lds_off(Qs), Qb + (size_t)q0 * HDIM, 64, 64, HDIM, 4, 3);
        __builtin_amdgcn_s_wait_tensorcnt(0);
    }
    __syncthreads();

    const int row0 = wave * 16;
    const int half = lane >> 4;
    const int nl = lane & 15;
    const v16h a_lo = load_a_frag(Qs + row0 * LDA, LDA, lane);       // d 0..31
    const v16h a_hi = load_a_frag(Qs + row0 * LDA + 32, LDA, lane);  // d 32..63

    float row_max[8], row_sum[8];
    v8f o[4] = {v8f{}, v8f{}, v8f{}, v8f{}};
#pragma unroll
    for (int v = 0; v < 8; ++v) { row_max[v] = -3.0e38f; row_sum[v] = 0.0f; }

    for (int k0 = 0; k0 <= q0; k0 += 64) {
        __syncthreads();   // previous tile fully consumed
        if (wave == 0) {
            tdm_load_2d_f16(lds_off(Ks), Kb + (size_t)k0 * HDIM, 64, 64, HDIM, 4, 3);
            tdm_load_2d_f16(lds_off(Vs), Vb + (size_t)k0 * HDIM, 64, 64, HDIM, 4, 3);
            __builtin_amdgcn_s_wait_tensorcnt(0);
        }
        __syncthreads();

        const bool diag = (k0 == q0);
        float p[4][8];
#pragma unroll
        for (int j = 0; j < 4; ++j) {
            v16h bl  = load_b_fragT(Ks + (j * 16) * LDA, LDA, lane);
            v16h bh2 = load_b_fragT(Ks + (j * 16) * LDA + 32, LDA, lane);
            v8f c = {};
            c = wmma16(a_lo, bl, c);
            c = wmma16(a_hi, bh2, c);
#pragma unroll
            for (int v = 0; v < 8; ++v) {
                float l = c[v] * 0.125f;   // 1/sqrt(64)
                if (diag && (k0 + j * 16 + nl) > (q0 + row0 + v + 8 * half))
                    l = -3.0e38f;
                p[j][v] = l;
            }
        }
        // online softmax per row (rows live in one 16-lane half)
#pragma unroll
        for (int v = 0; v < 8; ++v) {
            float m = fmaxf(fmaxf(p[0][v], p[1][v]), fmaxf(p[2][v], p[3][v]));
            m = fmaxf(m, __shfl_xor(m, 1, 32));
            m = fmaxf(m, __shfl_xor(m, 2, 32));
            m = fmaxf(m, __shfl_xor(m, 4, 32));
            m = fmaxf(m, __shfl_xor(m, 8, 32));
            const float nm = fmaxf(row_max[v], m);
            const float corr = __expf(row_max[v] - nm);
            row_max[v] = nm;
            float rs = 0.0f;
#pragma unroll
            for (int j = 0; j < 4; ++j) {
                p[j][v] = __expf(p[j][v] - nm);
                rs += p[j][v];
            }
            rs += __shfl_xor(rs, 1, 32);
            rs += __shfl_xor(rs, 2, 32);
            rs += __shfl_xor(rs, 4, 32);
            rs += __shfl_xor(rs, 8, 32);
            row_sum[v] = row_sum[v] * corr + rs;
#pragma unroll
            for (int f = 0; f < 4; ++f) o[f][v] *= corr;
        }
        // stage P (f16) as A-matrix: rows = q, cols = key
        unsigned short* Pw = Ps + wave * 16 * LDA;
#pragma unroll
        for (int j = 0; j < 4; ++j)
#pragma unroll
            for (int v = 0; v < 8; ++v)
                Pw[(v + 8 * half) * LDA + j * 16 + nl] = f2h(p[j][v]);

        const v16h pa_lo = load_a_frag(Pw, LDA, lane);       // keys 0..31
        const v16h pa_hi = load_a_frag(Pw + 32, LDA, lane);  // keys 32..63
#pragma unroll
        for (int f = 0; f < 4; ++f) {
            v16h vb_lo = load_b_fragN(Vs + f * 16, LDA, lane);
            v16h vb_hi = load_b_fragN(Vs + 32 * LDA + f * 16, LDA, lane);
            o[f] = wmma16(pa_lo, vb_lo, o[f]);
            o[f] = wmma16(pa_hi, vb_hi, o[f]);
        }
    }

    // write context, merged heads: CTX[b, s, h*64 + d] (f16)
    const int b = bh >> 4;
    const int h = bh & 15;
#pragma unroll
    for (int v = 0; v < 8; ++v) {
        const int s = q0 + row0 + v + 8 * half;
        const float inv = 1.0f / row_sum[v];
#pragma unroll
        for (int f = 0; f < 4; ++f) {
            const int e = h * HDIM + f * 16 + nl;
            CTX[((size_t)(b * S_LEN + s)) * E_DIM + e] = f2h(o[f][v] * inv);
        }
    }
}

// ================= Kernel 3: output projection =================
// out[m,n] (f32) = sum_k ctx[m,k] * Wo[n,k] + bo[n]. Both tiles via TDM.
__global__ __launch_bounds__(256) void proj_out_kernel(
    const unsigned short* __restrict__ CTX, const unsigned short* __restrict__ W16,
    const float* __restrict__ bias, float* __restrict__ Out) {
    __shared__ __align__(16) unsigned short As[64 * LDK];
    __shared__ __align__(16) unsigned short Bs[64 * LDK];
    const int m0 = blockIdx.x * 64;
    const int n0 = blockIdx.y * 64;
    const int t = threadIdx.x;
    const int lane = t & 31;
    const int wave = t >> 5;
    const int wm = wave >> 1;
    const int wn = wave & 1;

    v8f acc0 = {}, acc1 = {};
    for (int k0 = 0; k0 < E_DIM; k0 += 32) {
        __syncthreads();
        if (wave == 0) {
            tdm_load_2d_f16(lds_off(As), CTX + (size_t)m0 * E_DIM + k0,
                            32, 64, E_DIM, 3, 3);
            tdm_load_2d_f16(lds_off(Bs), W16 + (size_t)n0 * E_DIM + k0,
                            32, 64, E_DIM, 3, 3);
            __builtin_amdgcn_s_wait_tensorcnt(0);
        }
        __syncthreads();
        v16h a  = load_a_frag(As + (wm * 16) * LDK, LDK, lane);
        v16h b0 = load_b_fragT(Bs + (wn * 32) * LDK, LDK, lane);
        v16h b1 = load_b_fragT(Bs + (wn * 32 + 16) * LDK, LDK, lane);
        acc0 = wmma16(a, b0, acc0);
        acc1 = wmma16(a, b1, acc1);
    }
    const int half = lane >> 4;
    const int nl = lane & 15;
#pragma unroll
    for (int v = 0; v < 8; ++v) {
        const int m = m0 + wm * 16 + v + 8 * half;
        {
            const int n = n0 + wn * 32 + nl;
            Out[(size_t)m * E_DIM + n] = acc0[v] + bias[n];
        }
        {
            const int n = n0 + wn * 32 + 16 + nl;
            Out[(size_t)m * E_DIM + n] = acc1[v] + bias[n];
        }
    }
}

// ================= launcher =================
extern "C" void kernel_launch(void* const* d_in, const int* in_sizes, int n_in,
                              void* d_out, int out_size, void* d_ws, size_t ws_size,
                              hipStream_t stream) {
    const float* query = (const float*)d_in[0];
    const float* key   = (const float*)d_in[1];
    const float* value = (const float*)d_in[2];
    const float* Wq = (const float*)d_in[3];
    const float* bq = (const float*)d_in[4];
    const float* Wk = (const float*)d_in[5];
    const float* bk = (const float*)d_in[6];
    const float* Wv = (const float*)d_in[7];
    const float* bv = (const float*)d_in[8];
    const float* Wo = (const float*)d_in[9];
    const float* bo = (const float*)d_in[10];
    float* out = (float*)d_out;

    unsigned short* ws = (unsigned short*)d_ws;
    const size_t per  = (size_t)MTOT * E_DIM;    // 4096*1024
    const size_t wper = (size_t)E_DIM * E_DIM;   // 1024*1024
    unsigned short* Qh   = ws;
    unsigned short* Kh   = ws + per;
    unsigned short* Vh   = ws + 2 * per;
    unsigned short* CTX  = ws + 3 * per;
    unsigned short* Wq16 = ws + 4 * per;
    unsigned short* Wk16 = Wq16 + wper;
    unsigned short* Wv16 = Wk16 + wper;
    unsigned short* Wo16 = Wv16 + wper;

    // weights fp32 -> f16
    const int n4 = (int)(wper / 4);
    dim3 gCvt((n4 + 255) / 256);
    cvt_f16_kernel<<<gCvt, 256, 0, stream>>>(Wq, Wq16, n4);
    cvt_f16_kernel<<<gCvt, 256, 0, stream>>>(Wk, Wk16, n4);
    cvt_f16_kernel<<<gCvt, 256, 0, stream>>>(Wv, Wv16, n4);
    cvt_f16_kernel<<<gCvt, 256, 0, stream>>>(Wo, Wo16, n4);

    dim3 gProj(MTOT / 64, E_DIM / 64);  // 64 x 16
    proj_qkv_kernel<<<gProj, 256, 0, stream>>>(query, Wq16, bq, Qh);
    proj_qkv_kernel<<<gProj, 256, 0, stream>>>(key,   Wk16, bk, Kh);
    proj_qkv_kernel<<<gProj, 256, 0, stream>>>(value, Wv16, bv, Vh);

    dim3 gAttn(S_LEN / 64, BATCH * HEADS);  // 32 x 32
    flash_attn_kernel<<<gAttn, 128, 0, stream>>>(Qh, Kh, Vh, CTX);

    proj_out_kernel<<<gProj, 256, 0, stream>>>(CTX, Wo16, bo, out);
}
